// Seq2SeqWithAttention_52450140618768
// MI455X (gfx1250) — matrix operand order repaired and checked by reading, not compiled, not run on hardware
//
#include <hip/hip_runtime.h>
#include <hip/hip_bf16.h>

typedef __attribute__((ext_vector_type(16))) _Float16 v16h;
typedef __attribute__((ext_vector_type(8)))  _Float16 v8h;
typedef __attribute__((ext_vector_type(8)))  float    v8f;

#define TT 768
#define BB 32
#define HH 128
#define NG 512              // 4*H (gate dim)
#define MTOT (TT*BB)        // 24576 rows
#define KP0 320             // padded K for layer-0 input dim (300 -> 320)

__device__ __forceinline__ float sigf(float x) { return 1.f / (1.f + __expf(-x)); }

// ---- CDNA5 async global->LDS copy (ASYNCcnt-tracked) ------------------------
__device__ __forceinline__ unsigned lds_off(const void* p) {
  return (unsigned)(unsigned long long)(__attribute__((address_space(3))) const char*)p;
}
__device__ __forceinline__ void async_ld16(unsigned lds_byte, const void* g) {
  asm volatile("global_load_async_to_lds_b128 %0, %1, off"
               :: "v"(lds_byte), "v"((unsigned long long)g)
               : "memory");
}
__device__ __forceinline__ void wait_async0() {
  asm volatile("s_wait_asynccnt 0x0" ::: "memory");
}
// copy 64KB tile (32x512 f32) global -> LDS; 256 threads x 16 x 16B
__device__ __forceinline__ void async_copy_tile(const float* g, float* l, int tid) {
  unsigned lbase = lds_off(l) + tid * 16;
  const char* gb = (const char*)g + tid * 16;
#pragma unroll
  for (int i = 0; i < 16; ++i)
    async_ld16(lbase + i * 4096, gb + i * 4096);
}

// ---- WMMA fragment loaders (contiguous k order, wave32) -----------------
// A: 16x32 f16 tile, row-major source with leading dim `ld`.
// lanes 0-15 : row = lane,     elems 0..15 = K 0..15
// lanes 16-31: row = lane-16,  elems 0..15 = K 16..31
__device__ __forceinline__ v16h load_a_frag(const _Float16* src, int ld, int lane) {
  int half = lane >> 4, r = lane & 15;
  const _Float16* p = src + (size_t)r * ld;
  v8h lo = *(const v8h*)(p + 16 * half);
  v8h hi = *(const v8h*)(p + 16 * half + 8);
  v16h a;
#pragma unroll
  for (int e = 0; e < 8; ++e) { a[e] = lo[e]; a[8 + e] = hi[e]; }
  return a;
}

// B: 32x16 tile where B[k][n] = W[n][k], W row-major with leading dim `ld`.
// lane column n = lane&15; lanes 0-15 hold K 0..15, lanes 16-31 hold K 16..31.
// Caller passes w = W + n0*ld + k0 (n0,k0 tile origin).
__device__ __forceinline__ v16h load_b_frag(const _Float16* w, int ld, int lane) {
  int half = lane >> 4, n = lane & 15;
  return *(const v16h*)(w + (size_t)n * ld + 16 * half);
}

// ---- packing / conversion ---------------------------------------------------
// embeddings [B][T][300] f32 -> x16 [T][B][KP0] f16 (zero-padded K)
__global__ void k_pack_x(const float* __restrict__ src, _Float16* __restrict__ dst) {
  size_t i = (size_t)blockIdx.x * 256 + threadIdx.x;
  const size_t total = (size_t)MTOT * KP0;
  if (i >= total) return;
  int kp = (int)(i % KP0);
  size_t m = i / KP0;
  int b = (int)(m % BB), t = (int)(m / BB);
  dst[i] = (kp < 300) ? (_Float16)src[((size_t)b * TT + t) * 300 + kp] : (_Float16)0.f;
}

// generic f32 [N][ldsrc] (cols koff..koff+ktake) -> f16 [N][kp] zero-padded
__global__ void k_pack_w(const float* __restrict__ src, _Float16* __restrict__ dst,
                         int N, int ldsrc, int koff, int ktake, int kp) {
  long i = (long)blockIdx.x * 256 + threadIdx.x;
  long total = (long)N * kp;
  if (i >= total) return;
  int k = (int)(i % kp);
  long n = i / kp;
  dst[i] = (k < ktake) ? (_Float16)src[n * (long)ldsrc + koff + k] : (_Float16)0.f;
}

// ---- input-projection GEMM: C[M][512] = A16[M][K] @ W16[512][K]^T + b0 + b1
// grid: (8, M/128), block 256 (8 waves). wave -> 16x64 strip, 4 WMMA tiles.
__global__ void __launch_bounds__(256) k_gemm_bias(
    const _Float16* __restrict__ A, const _Float16* __restrict__ W,
    const float* __restrict__ b0, const float* __restrict__ b1,
    float* __restrict__ C, int K) {
  int lane = threadIdx.x & 31, wave = threadIdx.x >> 5;
  int m0 = blockIdx.y * 128 + wave * 16;
  int n0 = blockIdx.x * 64;
  v8f zero = {};
  v8f acc[4];
#pragma unroll
  for (int t = 0; t < 4; ++t) acc[t] = zero;
  for (int k0 = 0; k0 < K; k0 += 32) {
    v16h a = load_a_frag(A + (size_t)m0 * K + k0, K, lane);
#pragma unroll
    for (int t = 0; t < 4; ++t) {
      v16h b = load_b_frag(W + (size_t)(n0 + t * 16) * K + k0, K, lane);
      acc[t] = __builtin_amdgcn_wmma_f32_16x16x32_f16(false, a, false, b,
                                                      (short)0, acc[t], false, false);
    }
  }
  int half = lane >> 4, col = lane & 15;
#pragma unroll
  for (int t = 0; t < 4; ++t) {
    int n = n0 + t * 16 + col;
    float bias = b0[n] + b1[n];
#pragma unroll
    for (int v = 0; v < 8; ++v) {
      int m = m0 + v + 8 * half;
      C[(size_t)m * NG + n] = acc[t][v] + bias;
    }
  }
}

// ---- LSTM scan: one workgroup per (model,direction) ------------------------
// gates[32][512] = h[32][128] @ Whh^T + xproj[t]; full LSTM cell in-register.
// Wave w owns gate columns j in [16w,16w+16); its 4 N-tiles are the SAME j
// slice of gates i/f/g/o, so the cell never leaves the C-fragment layout.
// The per-step 64KB xproj tile is double-buffered in LDS via async copies.
struct ScanArgs {
  const float* xproj[4];   // [T][32][512] f32
  const float* whh[4];     // [512][128]  f32 (converted to B-frags in-kernel)
  float* out[4];           // [T][32][256] f32, feature offset pre-applied
  int rev[4];
};

__global__ void __launch_bounds__(256) k_scan(ScanArgs args, int T) {
  __shared__ _Float16 sH[BB * HH];       // 8 KB h-state (f16 for WMMA A)
  __shared__ float xbuf[2][BB * NG];     // 2 x 64 KB xproj tiles
  int blk = blockIdx.x;
  const float* xproj = args.xproj[blk];
  const float* whh   = args.whh[blk];
  float* out         = args.out[blk];
  int rev            = args.rev[blk];

  int tid = threadIdx.x, lane = tid & 31, wave = tid >> 5;
  int col = lane & 15, half = lane >> 4;

  // Kick off the async copy of the first tile immediately.
  int t0 = rev ? (T - 1) : 0;
  async_copy_tile(xproj + (size_t)t0 * (BB * NG), &xbuf[0][0], tid);

  // Pre-build all 16 Whh^T B-fragments in registers (resident for all T steps)
  v16h Bf[4][4];
#pragma unroll
  for (int nt = 0; nt < 4; ++nt) {
    int n = nt * HH + wave * 16 + col;          // gate-nt, column j = 16w+col
    const float* wr = whh + (size_t)n * HH;
#pragma unroll
    for (int kc = 0; kc < 4; ++kc) {
      v16h bb;
#pragma unroll
      for (int e = 0; e < 16; ++e) bb[e] = (_Float16)wr[kc * 32 + 16 * half + e];
      Bf[nt][kc] = bb;
    }
  }
  for (int i = tid; i < BB * HH; i += 256) sH[i] = (_Float16)0.f;
  v8f zero = {};
  v8f cs[2];                                     // cell state, rows mt*16+v+8*half
  cs[0] = zero; cs[1] = zero;
  wait_async0();                                 // first tile landed (this wave)
  __syncthreads();                               // ... and all waves

  int j = wave * 16 + col;
  for (int s = 0; s < T; ++s) {
    int t = rev ? (T - 1 - s) : s;
    if (s + 1 < T) {                             // uniform branch: EXEC stays full
      int tn = rev ? (T - 2 - s) : (s + 1);
      async_copy_tile(xproj + (size_t)tn * (BB * NG), &xbuf[(s + 1) & 1][0], tid);
    }
    v8f acc[2][4];
#pragma unroll
    for (int mt = 0; mt < 2; ++mt)
#pragma unroll
      for (int nt = 0; nt < 4; ++nt) acc[mt][nt] = zero;
#pragma unroll
    for (int kc = 0; kc < 4; ++kc) {
      v16h a0 = load_a_frag(sH + kc * 32, HH, lane);            // rows 0..15
      v16h a1 = load_a_frag(sH + 16 * HH + kc * 32, HH, lane);  // rows 16..31
#pragma unroll
      for (int nt = 0; nt < 4; ++nt) {
        acc[0][nt] = __builtin_amdgcn_wmma_f32_16x16x32_f16(false, a0, false, Bf[nt][kc],
                                                            (short)0, acc[0][nt], false, false);
        acc[1][nt] = __builtin_amdgcn_wmma_f32_16x16x32_f16(false, a1, false, Bf[nt][kc],
                                                            (short)0, acc[1][nt], false, false);
      }
    }
    __syncthreads();   // all A reads of sH done before overwrite
    const float* xb = &xbuf[s & 1][0];
    float* orow = out + (size_t)t * (BB * 2 * HH);
#pragma unroll
    for (int mt = 0; mt < 2; ++mt) {
#pragma unroll
      for (int v = 0; v < 8; ++v) {
        int b = mt * 16 + v + 8 * half;
        const float* xr = xb + b * NG;
        float gi = acc[mt][0][v] + xr[j];
        float gf = acc[mt][1][v] + xr[HH + j];
        float gg = acc[mt][2][v] + xr[2 * HH + j];
        float go = acc[mt][3][v] + xr[3 * HH + j];
        float c  = sigf(gf) * cs[mt][v] + sigf(gi) * tanhf(gg);
        float h  = sigf(go) * tanhf(c);
        cs[mt][v] = c;
        sH[b * HH + j] = (_Float16)h;
        orow[(size_t)b * (2 * HH) + j] = h;
      }
    }
    wait_async0();     // next tile landed (this wave's copies)
    __syncthreads();   // h-state + everyone's tile copies visible
  }
}

// ---- collapsed attention: p = softmax_t(w_enc . enc[t,b,:]); ctx = sum p*enc
// then hc[b] = ctx[b] @ w1[:, :256]^T. One block per batch b.
__global__ void __launch_bounds__(256) k_ctx(const float* __restrict__ enc,
                                             const float* __restrict__ w_att,
                                             const float* __restrict__ w1,
                                             float* __restrict__ hc, int T) {
  __shared__ float sw[256];
  __shared__ float sp[TT];
  __shared__ float red[256];
  __shared__ float sc[256];
  int b = blockIdx.x, tid = threadIdx.x;
  sw[tid] = w_att[256 + tid];                     // w_enc part
  __syncthreads();
  for (int t = tid; t < T; t += 256) {
    const float* row = enc + ((size_t)t * BB + b) * (2 * HH);
    float s = 0.f;
    for (int f = 0; f < 256; ++f) s += row[f] * sw[f];
    sp[t] = s;                                    // b_att/s_dec cancel in softmax
  }
  __syncthreads();
  float m = -1e30f;
  for (int t = tid; t < T; t += 256) m = fmaxf(m, sp[t]);
  red[tid] = m; __syncthreads();
  for (int o = 128; o > 0; o >>= 1) { if (tid < o) red[tid] = fmaxf(red[tid], red[tid + o]); __syncthreads(); }
  float mx = red[0]; __syncthreads();
  float ssum = 0.f;
  for (int t = tid; t < T; t += 256) { float e = __expf(sp[t] - mx); sp[t] = e; ssum += e; }
  red[tid] = ssum; __syncthreads();
  for (int o = 128; o > 0; o >>= 1) { if (tid < o) red[tid] += red[tid + o]; __syncthreads(); }
  float inv = 1.f / red[0];
  __syncthreads();
  float a = 0.f;                                  // thread owns feature tid
  for (int t = 0; t < T; ++t) a += sp[t] * enc[((size_t)t * BB + b) * (2 * HH) + tid];
  sc[tid] = a * inv;
  __syncthreads();
  if (tid < HH) {
    const float* wr = w1 + (size_t)tid * NG;      // w1 row, context half
    float acc = 0.f;
    for (int k = 0; k < 256; ++k) acc += sc[k] * wr[k];
    hc[b * HH + tid] = acc;
  }
}

// ---- final FC: hpre[m][128] = dec16[m][256] @ w1b16[128][256]^T + b1 + hc[b]
__global__ void __launch_bounds__(256) k_fc_dec(const _Float16* __restrict__ A,
                                                const _Float16* __restrict__ W,
                                                const float* __restrict__ b1v,
                                                const float* __restrict__ hc,
                                                float* __restrict__ C) {
  const int K = 256;
  int lane = threadIdx.x & 31, wave = threadIdx.x >> 5;
  int m0 = blockIdx.x * 64 + (wave >> 1) * 16;
  int n0 = (wave & 1) * 64;
  v8f zero = {};
  v8f acc[4];
#pragma unroll
  for (int t = 0; t < 4; ++t) acc[t] = zero;
  for (int k0 = 0; k0 < K; k0 += 32) {
    v16h a = load_a_frag(A + (size_t)m0 * K + k0, K, lane);
#pragma unroll
    for (int t = 0; t < 4; ++t) {
      v16h b = load_b_frag(W + (size_t)(n0 + t * 16) * K + k0, K, lane);
      acc[t] = __builtin_amdgcn_wmma_f32_16x16x32_f16(false, a, false, b,
                                                      (short)0, acc[t], false, false);
    }
  }
  int half = lane >> 4, col = lane & 15;
#pragma unroll
  for (int t = 0; t < 4; ++t) {
    int n = n0 + t * 16 + col;
#pragma unroll
    for (int v = 0; v < 8; ++v) {
      int m = m0 + v + 8 * half;
      int b = m & (BB - 1);
      C[(size_t)m * HH + n] = acc[t][v] + b1v[n] + hc[b * HH + n];
    }
  }
}

// ---- batchnorm stats (deterministic tree reduction, no atomics) ------------
__global__ void k_bn_part(const float* __restrict__ h, float* __restrict__ part) {
  int f = threadIdx.x;                 // 128 threads
  int r0 = blockIdx.x * 256;           // 96 blocks x 256 rows
  float s = 0.f, s2 = 0.f;
  for (int r = 0; r < 256; ++r) {
    float v = h[(size_t)(r0 + r) * HH + f];
    s += v; s2 += v * v;
  }
  part[(blockIdx.x * 2 + 0) * HH + f] = s;
  part[(blockIdx.x * 2 + 1) * HH + f] = s2;
}

__global__ void k_bn_final(const float* __restrict__ part, const float* __restrict__ gamma,
                           const float* __restrict__ beta, float* __restrict__ scale,
                           float* __restrict__ shift) {
  int f = threadIdx.x;
  float s = 0.f, s2 = 0.f;
  for (int p = 0; p < 96; ++p) { s += part[(p * 2) * HH + f]; s2 += part[(p * 2 + 1) * HH + f]; }
  float mean = s / (float)MTOT;
  float var  = s2 / (float)MTOT - mean * mean;
  float sc = gamma[f] * rsqrtf(var + 1e-5f);
  scale[f] = sc;
  shift[f] = beta[f] - mean * sc;
}

// ---- BN + relu + final linear + sigmoid ------------------------------------
__global__ void __launch_bounds__(256) k_final(const float* __restrict__ h,
                                               const float* __restrict__ scale,
                                               const float* __restrict__ shift,
                                               const float* __restrict__ w2,
                                               const float* __restrict__ b2,
                                               float* __restrict__ out) {
  __shared__ float sw[HH], ssc[HH], ssh[HH];
  int tid = threadIdx.x;
  if (tid < HH) { sw[tid] = w2[tid]; ssc[tid] = scale[tid]; ssh[tid] = shift[tid]; }
  __syncthreads();
  int m = blockIdx.x * 256 + tid;
  if (m < MTOT) {
    const float* row = h + (size_t)m * HH;
    float a = 0.f;
    for (int f = 0; f < HH; ++f) {
      float v = row[f] * ssc[f] + ssh[f];
      v = fmaxf(v, 0.f);
      a += v * sw[f];
    }
    float o = 1.f / (1.f + __expf(-(a + b2[0])));
    int b = m & (BB - 1), t = m >> 5;
    out[(size_t)b * TT + t] = o;                 // out.reshape(B, T)
  }
}

// ---- host side --------------------------------------------------------------
struct DirP { const float *Wih, *Whh, *bih, *bhh; };

extern "C" void kernel_launch(void* const* d_in, const int* in_sizes, int n_in,
                              void* d_out, int out_size, void* d_ws, size_t ws_size,
                              hipStream_t stream) {
  (void)out_size; (void)ws_size;
  const float* emb = (const float*)d_in[0];
  // Tail params (fixed top-level order): w_att,b_att,w1,b1,gamma,beta,w2,b2
  const float* w_att = (const float*)d_in[n_in - 8];
  const float* w1    = (const float*)d_in[n_in - 6];
  const float* b1    = (const float*)d_in[n_in - 5];
  const float* gamma = (const float*)d_in[n_in - 4];
  const float* beta  = (const float*)d_in[n_in - 3];
  const float* w2    = (const float*)d_in[n_in - 2];
  const float* b2    = (const float*)d_in[n_in - 1];
  float* out = (float*)d_out;

  // JAX pytree flatten: dict keys sorted -> per model: [l0.bwd, l0.fwd, l1.bwd, l1.fwd],
  // each group 4 tensors; identify Wih/Whh/biases by size (bih/bhh only summed).
  const int per_model = (n_in - 10) / 2;   // 16
  auto getdir = [&](int model, int layer, int fwd) -> DirP {
    int g = 2 + model * per_model + layer * 8 + (fwd ? 4 : 0);
    DirP p{};
    const float* bias[2] = {nullptr, nullptr};
    int nb = 0;
    int wihN = (layer == 0) ? (NG * 300) : (NG * 256);
    for (int i = 0; i < 4; ++i) {
      int sz = in_sizes[g + i];
      const float* ptr = (const float*)d_in[g + i];
      if (sz == NG * HH)      p.Whh = ptr;
      else if (sz == wihN)    p.Wih = ptr;
      else if (nb < 2)        bias[nb++] = ptr;
    }
    p.bih = bias[0]; p.bhh = bias[1];
    return p;
  };

  // workspace carve-up (256B aligned)
  char* base = (char*)d_ws;
  size_t off = 0;
  auto carve = [&](size_t bytes) -> char* {
    char* p = base + off;
    off += (bytes + 255) & ~(size_t)255;
    return p;
  };
  _Float16* x16      = (_Float16*)carve((size_t)MTOT * KP0 * 2);
  _Float16* wih16[4];
  for (int d = 0; d < 4; ++d) wih16[d] = (_Float16*)carve((size_t)NG * KP0 * 2);
  float* xproj[4];
  for (int d = 0; d < 4; ++d) xproj[d] = (float*)carve((size_t)MTOT * NG * 4);
  float* seq0e = (float*)carve((size_t)MTOT * 256 * 4);
  float* seq0d = (float*)carve((size_t)MTOT * 256 * 4);
  float* seq1e = (float*)carve((size_t)MTOT * 256 * 4);
  float* seq1d = (float*)carve((size_t)MTOT * 256 * 4);
  _Float16* a16e = (_Float16*)carve((size_t)MTOT * 256 * 2);
  _Float16* a16d = (_Float16*)carve((size_t)MTOT * 256 * 2);
  _Float16* w1b16 = (_Float16*)carve((size_t)HH * 256 * 2);
  float* hc     = (float*)carve((size_t)BB * HH * 4);
  float* hpre   = (float*)carve((size_t)MTOT * HH * 4);
  float* bnpart = (float*)carve((size_t)96 * 2 * HH * 4);
  float* bnscale = (float*)carve(HH * 4);
  float* bnshift = (float*)carve(HH * 4);

  DirP p0[4], p1[4];
  for (int d = 0; d < 4; ++d) {
    p0[d] = getdir(d >> 1, 0, (d & 1) == 0);
    p1[d] = getdir(d >> 1, 1, (d & 1) == 0);
  }

  // 1) pack embeddings -> [T][B][320] f16 ; Wih layer0 -> f16 padded
  k_pack_x<<<(MTOT * KP0) / 256, 256, 0, stream>>>(emb, x16);
  for (int d = 0; d < 4; ++d)
    k_pack_w<<<(NG * KP0) / 256, 256, 0, stream>>>(p0[d].Wih, wih16[d], NG, 300, 0, 300, KP0);

  // 2) layer-0 input projections (WMMA GEMM) + 3) layer-0 scans (4 concurrent WGs)
  for (int d = 0; d < 4; ++d)
    k_gemm_bias<<<dim3(8, MTOT / 128), 256, 0, stream>>>(x16, wih16[d], p0[d].bih, p0[d].bhh,
                                                         xproj[d], KP0);
  ScanArgs s0;
  for (int d = 0; d < 4; ++d) {
    s0.xproj[d] = xproj[d];
    s0.whh[d]   = p0[d].Whh;
    s0.out[d]   = ((d < 2) ? seq0e : seq0d) + ((d & 1) ? HH : 0);
    s0.rev[d]   = d & 1;
  }
  k_scan<<<4, 256, 0, stream>>>(s0, TT);

  // 4) layer-1: convert layer-0 outputs + Wih, project, scan
  k_pack_w<<<(MTOT * 256) / 256, 256, 0, stream>>>(seq0e, a16e, MTOT, 256, 0, 256, 256);
  k_pack_w<<<(MTOT * 256) / 256, 256, 0, stream>>>(seq0d, a16d, MTOT, 256, 0, 256, 256);
  for (int d = 0; d < 4; ++d)
    k_pack_w<<<(NG * 256) / 256, 256, 0, stream>>>(p1[d].Wih, wih16[d], NG, 256, 0, 256, 256);
  for (int d = 0; d < 4; ++d)
    k_gemm_bias<<<dim3(8, MTOT / 128), 256, 0, stream>>>((d < 2) ? a16e : a16d, wih16[d],
                                                         p1[d].bih, p1[d].bhh, xproj[d], 256);
  ScanArgs s1;
  for (int d = 0; d < 4; ++d) {
    s1.xproj[d] = xproj[d];
    s1.whh[d]   = p1[d].Whh;
    s1.out[d]   = ((d < 2) ? seq1e : seq1d) + ((d & 1) ? HH : 0);
    s1.rev[d]   = d & 1;
  }
  k_scan<<<4, 256, 0, stream>>>(s1, TT);

  // 5) collapsed attention -> hc[b][128]
  k_ctx<<<BB, 256, 0, stream>>>(seq1e, w_att, w1, hc, TT);

  // 6) final FC over dec (WMMA) + batchnorm + head
  k_pack_w<<<(MTOT * 256) / 256, 256, 0, stream>>>(seq1d, a16d, MTOT, 256, 0, 256, 256);
  k_pack_w<<<(HH * 256) / 256, 256, 0, stream>>>(w1, w1b16, HH, NG, 256, 256, 256);
  k_fc_dec<<<MTOT / 64, 256, 0, stream>>>(a16d, w1b16, b1, hc, hpre);
  k_bn_part<<<96, 128, 0, stream>>>(hpre, bnpart);
  k_bn_final<<<1, 128, 0, stream>>>(bnpart, gamma, beta, bnscale, bnshift);
  k_final<<<MTOT / 256, 256, 0, stream>>>(hpre, bnscale, bnshift, w2, b2, out);
}
